// GiddAttention_84567906058330
// MI455X (gfx1250) — matrix operand order repaired and checked by reading, not compiled, not run on hardware
//
#include <hip/hip_runtime.h>
#include <math.h>

// ---------------------------------------------------------------------------
// GiddAttention for MI455X (gfx1250): bf16 WMMA everywhere, flash attention,
// async global->LDS staging for the attention K tiles.
// ---------------------------------------------------------------------------

typedef __bf16 bf16;
typedef __attribute__((ext_vector_type(16))) __bf16 v16bf;
typedef __attribute__((ext_vector_type(8)))  __bf16 bf16x8;
typedef __attribute__((ext_vector_type(8)))  float  v8f;
typedef __attribute__((ext_vector_type(4)))  int    i32x4;

#define DEVFN static __device__ __forceinline__

constexpr int B_   = 2;
constexpr int L_   = 2048;
constexpr int HID_ = 2048;
constexpr int H_   = 16;
constexpr int D_   = 128;
constexpr int BL_  = B_ * L_;

constexpr float FAN_IN = 0.022097086912079608f;   // 1/sqrt(2048)
constexpr float ASCALE = 0.08838834764831845f;    // 1/sqrt(128)
constexpr float EPS_   = 1e-6f;
constexpr float CAP_   = 30.0f;
constexpr float NEGF   = -3.4028234663852886e38f; // finfo(f32).min

DEVFN v8f zero8() {
  v8f z;
#pragma unroll
  for (int i = 0; i < 8; ++i) z[i] = 0.0f;
  return z;
}

// Build a 16-element bf16 fragment from two 16-byte chunks.
DEVFN v16bf load_frag(const bf16* p0, const bf16* p1) {
  union { v16bf v; bf16x8 h[2]; } u;
  u.h[0] = *(const bf16x8*)p0;
  u.h[1] = *(const bf16x8*)p1;
  return u.v;
}

DEVFN v8f wmma_bf16(v16bf a, v16bf b, v8f c) {
  return __builtin_amdgcn_wmma_f32_16x16x32_bf16(
      /*neg_a=*/false, a, /*neg_b=*/false, b,
      /*c_mod=*/(short)0, c, /*reuse_a=*/false, /*reuse_b=*/false);
}

// ---- CDNA5 async global->LDS copy (ASYNCcnt path), with safe fallback ------
DEVFN void async_ld16(const bf16* g, bf16* l) {
#if __has_builtin(__builtin_amdgcn_global_load_async_to_lds_b128)
  __builtin_amdgcn_global_load_async_to_lds_b128(
      (__attribute__((address_space(1))) i32x4*)g,
      (__attribute__((address_space(3))) i32x4*)l, /*offset=*/0, /*cpol=*/0);
#else
  *(bf16x8*)l = *(const bf16x8*)g;  // fallback: VGPR round trip
#endif
}

DEVFN void wait_async() {
#if __has_builtin(__builtin_amdgcn_global_load_async_to_lds_b128)
#if __has_builtin(__builtin_amdgcn_s_wait_asynccnt)
  __builtin_amdgcn_s_wait_asynccnt(0);
#else
  asm volatile("s_wait_asynccnt 0" ::: "memory");
#endif
#endif
}

// ---------------------------------------------------------------------------
// f32 -> bf16 conversion (grid-stride)
// ---------------------------------------------------------------------------
__global__ void cvt_bf16(const float* __restrict__ x, bf16* __restrict__ y, int n) {
  int i = blockIdx.x * blockDim.x + threadIdx.x;
  int stride = gridDim.x * blockDim.x;
  for (; i < n; i += stride) y[i] = (bf16)x[i];
}

// ---------------------------------------------------------------------------
// C[M,N] = (A[M,K] @ Bw[N,K]^T) * scale   (all bf16 in, OutT out)
// Wave tile 64x64 as 4x4 WMMA 16x16x32_bf16, fragments streamed from global.
// Grid: (N/256, M/128), block 256 (8 waves arranged 2Mx4N).
// ---------------------------------------------------------------------------
template <typename OutT>
__global__ __launch_bounds__(256)
void gemm_bf16_nt(const bf16* __restrict__ A, const bf16* __restrict__ Bw,
                  OutT* __restrict__ C, float scale) {
  constexpr int K = HID_, N = HID_;
  const int lane = threadIdx.x & 31;
  const int wave = threadIdx.x >> 5;
  const int hi   = lane >> 4;     // which 16-lane half
  const int col  = lane & 15;
  const int m0 = blockIdx.y * 128 + (wave >> 2) * 64;
  const int n0 = blockIdx.x * 256 + (wave & 3) * 64;

  v8f acc[4][4];
#pragma unroll
  for (int mi = 0; mi < 4; ++mi)
#pragma unroll
    for (int ni = 0; ni < 4; ++ni) acc[mi][ni] = zero8();

  for (int k0 = 0; k0 < K; k0 += 32) {
    v16bf a[4];
#pragma unroll
    for (int mi = 0; mi < 4; ++mi) {
      // A fragment (16x32): lane_lo K = {0..7,16..23}, lane_hi K = {8..15,24..31}
      const bf16* ap = A + (size_t)(m0 + mi * 16 + col) * K + k0 + hi * 8;
      a[mi] = load_frag(ap, ap + 16);
    }
#pragma unroll
    for (int ni = 0; ni < 4; ++ni) {
      // B fragment (32x16) from [n][k]-major Bw: lane_lo K = {0..7,16..23}, lane_hi K = {8..15,24..31}
      const bf16* bp = Bw + (size_t)(n0 + ni * 16 + col) * K + k0 + hi * 8;
      v16bf bfr = load_frag(bp, bp + 16);
#pragma unroll
      for (int mi = 0; mi < 4; ++mi)
        acc[mi][ni] = wmma_bf16(a[mi], bfr, acc[mi][ni]);
    }
  }

#pragma unroll
  for (int mi = 0; mi < 4; ++mi)
#pragma unroll
    for (int ni = 0; ni < 4; ++ni) {
      const int rowb = m0 + mi * 16 + hi * 8;  // C layout: VGPR r -> M = r (+8 for hi)
      const int c    = n0 + ni * 16 + col;
#pragma unroll
      for (int r = 0; r < 8; ++r)
        C[(size_t)(rowb + r) * N + c] = (OutT)(acc[mi][ni][r] * scale);
    }
}

// ---------------------------------------------------------------------------
// Per-token: RMS-norm(q,k over 2048) + neox RoPE + k/v bias + relayout to
// [B,H,L,D] bf16. Attention scale folded into Q.
// ---------------------------------------------------------------------------
__global__ __launch_bounds__(256)
void post_kernel(const bf16* __restrict__ Qb, const bf16* __restrict__ Kb,
                 const bf16* __restrict__ Vb,
                 const float* __restrict__ qw, const float* __restrict__ kw,
                 const float* __restrict__ kbias, const float* __restrict__ vbias,
                 const int* __restrict__ positions,
                 bf16* __restrict__ Qh, bf16* __restrict__ Kh, bf16* __restrict__ Vh) {
  const int t = blockIdx.x;           // token index in [0, B*L)
  const int b = t / L_, l = t % L_;
  const int tid = threadIdx.x;
  const bf16* qrow = Qb + (size_t)t * HID_;
  const bf16* krow = Kb + (size_t)t * HID_;
  const bf16* vrow = Vb + (size_t)t * HID_;

  float sq = 0.f, sk = 0.f;
  for (int i = tid; i < HID_; i += 256) {
    float qv = (float)qrow[i]; sq += qv * qv;
    float kv = (float)krow[i]; sk += kv * kv;
  }
#pragma unroll
  for (int off = 16; off; off >>= 1) {
    sq += __shfl_xor(sq, off, 32);
    sk += __shfl_xor(sk, off, 32);
  }
  __shared__ float redq[8], redk[8];
  if ((tid & 31) == 0) { redq[tid >> 5] = sq; redk[tid >> 5] = sk; }
  __syncthreads();
  float tq = 0.f, tk = 0.f;
#pragma unroll
  for (int w = 0; w < 8; ++w) { tq += redq[w]; tk += redk[w]; }
  const float rq = rsqrtf(tq * (1.0f / HID_) + EPS_);
  const float rk = rsqrtf(tk * (1.0f / HID_) + EPS_);
  const float pos = (float)positions[t];

  for (int pi = tid; pi < HID_ / 2; pi += 256) {
    const int hh = pi >> 6;       // head
    const int j  = pi & 63;       // rotary pair index
    const int i1 = hh * D_ + j;
    const int i2 = i1 + 64;
    // inv_freq = 10000^(-j/64) = exp2(-j * log2(1e4) / 64)
    const float inv = exp2f((float)j * (-13.287712379549449f / 64.0f));
    float sn, cs;
    sincosf(pos * inv, &sn, &cs);

    const float q1 = (1.0f + qw[i1]) * (float)qrow[i1] * rq;
    const float q2 = (1.0f + qw[i2]) * (float)qrow[i2] * rq;
    const float k1 = (1.0f + kw[i1]) * (float)krow[i1] * rk;
    const float k2 = (1.0f + kw[i2]) * (float)krow[i2] * rk;

    const float qo1 = (q1 * cs - q2 * sn) * ASCALE;
    const float qo2 = (q2 * cs + q1 * sn) * ASCALE;
    const float ko1 = k1 * cs - k2 * sn + kbias[i1];
    const float ko2 = k2 * cs + k1 * sn + kbias[i2];
    const float vo1 = (float)vrow[i1] + vbias[i1];
    const float vo2 = (float)vrow[i2] + vbias[i2];

    const size_t o1 = ((size_t)(b * H_ + hh) * L_ + l) * D_ + j;
    Qh[o1] = (bf16)qo1; Qh[o1 + 64] = (bf16)qo2;
    Kh[o1] = (bf16)ko1; Kh[o1 + 64] = (bf16)ko2;
    Vh[o1] = (bf16)vo1; Vh[o1 + 64] = (bf16)vo2;
  }
}

// ---------------------------------------------------------------------------
// Flash attention: one block per (128-query tile, h, b); 8 waves x 16 queries.
// Key blocks of 64. K tile async-staged in LDS (shared by all 8 waves),
// V tile transposed into LDS, QK^T and PV via WMMA bf16,
// softcap + mask + online softmax.
// ---------------------------------------------------------------------------
__global__ __launch_bounds__(256)
void attn_kernel(const bf16* __restrict__ Qh, const bf16* __restrict__ Kh,
                 const bf16* __restrict__ Vh, const unsigned char* __restrict__ mask,
                 bf16* __restrict__ Ab) {
  const int lane = threadIdx.x & 31;
  const int wave = threadIdx.x >> 5;
  const int hi   = lane >> 4;
  const int col  = lane & 15;
  const int b = blockIdx.z, h = blockIdx.y;
  const int qrow0 = blockIdx.x * 128 + wave * 16;
  const size_t bh = (size_t)(b * H_ + h) * L_ * D_;

  __shared__ bf16 lK[64 * 136];        // K tile: [key][d], stride 136 (pad)
  __shared__ bf16 lVt[128 * 72];       // V^T tile: [d][key], stride 72 (pad)
  __shared__ bf16 lP[8 * 16 * 72];     // per-wave P scratch: [16][64], stride 72
  bf16* lPw = lP + wave * (16 * 72);

  // Q A-fragments (resident across all key blocks), D=128 -> 4 chunks of 32.
  v16bf aq[4];
  {
    const bf16* qp = Qh + bh + (size_t)(qrow0 + col) * D_;
#pragma unroll
    for (int kc = 0; kc < 4; ++kc) {
      const int dof = kc * 32 + hi * 8;
      aq[kc] = load_frag(qp + dof, qp + dof + 16);
    }
  }

  v8f o[8];
  float rm[8], rl[8];
#pragma unroll
  for (int di = 0; di < 8; ++di) o[di] = zero8();
#pragma unroll
  for (int r = 0; r < 8; ++r) { rm[r] = -__builtin_inff(); rl[r] = 0.0f; }

  const unsigned char* mbase = mask + (size_t)b * L_ * L_;

  for (int kb0 = 0; kb0 < L_; kb0 += 64) {
    __syncthreads();   // previous block's LDS reads complete

    { // async-stage K block into LDS: lK[key][d], 64 rows x 256B, 64B/thread
      const int key   = threadIdx.x >> 2;        // 0..63
      const int dbase = (threadIdx.x & 3) * 32;  // 0,32,64,96 (elements)
      const bf16* kg = Kh + bh + (size_t)(kb0 + key) * D_ + dbase;
      bf16* kl = lK + key * 136 + dbase;
#pragma unroll
      for (int c = 0; c < 4; ++c)
        async_ld16(kg + c * 8, kl + c * 8);
    }

    { // cooperative transpose-load of V block into LDS: lVt[d][key]
      const int key   = threadIdx.x >> 2;        // 0..63
      const int dbase = (threadIdx.x & 3) * 32;  // 0,32,64,96
      const bf16* vp = Vh + bh + (size_t)(kb0 + key) * D_ + dbase;
#pragma unroll
      for (int c = 0; c < 4; ++c) {
        bf16x8 v8 = *(const bf16x8*)(vp + c * 8);
#pragma unroll
        for (int e = 0; e < 8; ++e)
          lVt[(dbase + c * 8 + e) * 72 + key] = v8[e];
      }
    }
    wait_async();      // this wave's async copies landed in LDS
    __syncthreads();   // publish all waves' K/V tiles

    // ---- S = Q @ K^T for 4 key sub-tiles of 16 ----
    v8f s[4];
#pragma unroll
    for (int ni = 0; ni < 4; ++ni) s[ni] = zero8();
#pragma unroll
    for (int ni = 0; ni < 4; ++ni) {
      const bf16* kp = lK + (ni * 16 + col) * 136;
#pragma unroll
      for (int kc = 0; kc < 4; ++kc) {
        const bf16* kpp = kp + kc * 32 + hi * 8;
        v16bf bk = load_frag(kpp, kpp + 16);
        s[ni] = wmma_bf16(aq[kc], bk, s[ni]);
      }
    }

    // ---- softcap + mask + online softmax (per-row, rows live per 16-lane half) ----
    float corr[8];
#pragma unroll
    for (int r = 0; r < 8; ++r) {
      const int qg = qrow0 + hi * 8 + r;
      const unsigned char* mr = mbase + (size_t)qg * L_ + kb0;
      float vmax = NEGF;
#pragma unroll
      for (int ni = 0; ni < 4; ++ni) {
        float sc = s[ni][r];
        sc = CAP_ * tanhf(sc * (1.0f / CAP_));
        sc = mr[ni * 16 + col] ? sc : NEGF;
        s[ni][r] = sc;
        vmax = fmaxf(vmax, sc);
      }
#pragma unroll
      for (int off = 1; off < 16; off <<= 1)
        vmax = fmaxf(vmax, __shfl_xor(vmax, off, 16));
      const float mnew = fmaxf(rm[r], vmax);
      const float c = __expf(rm[r] - mnew);   // rm=-inf -> 0 on first block
      float psum = 0.0f;
#pragma unroll
      for (int ni = 0; ni < 4; ++ni) {
        const float sc = s[ni][r];
        const float p = (sc > -1e37f) ? __expf(sc - mnew) : 0.0f;
        lPw[(hi * 8 + r) * 72 + ni * 16 + col] = (bf16)p;
        psum += p;
      }
#pragma unroll
      for (int off = 1; off < 16; off <<= 1)
        psum += __shfl_xor(psum, off, 16);
      rl[r] = rl[r] * c + psum;
      rm[r] = mnew;
      corr[r] = c;
    }

    // Wave-internal LDS RAW (cross-lane): wait for DS stores before frag loads.
    asm volatile("s_wait_dscnt 0" ::: "memory");

    // ---- P A-fragments (16x64 -> 2 chunks of 32) ----
    v16bf pf[2];
#pragma unroll
    for (int kc = 0; kc < 2; ++kc) {
      const bf16* pp = lPw + col * 72 + kc * 32 + hi * 8;
      pf[kc] = load_frag(pp, pp + 16);
    }

    // ---- O = O*corr + P @ V ----
#pragma unroll
    for (int di = 0; di < 8; ++di) {
#pragma unroll
      for (int r = 0; r < 8; ++r) o[di][r] *= corr[r];
#pragma unroll
      for (int kc = 0; kc < 2; ++kc) {
        const bf16* vp = &lVt[(di * 16 + col) * 72 + kc * 32 + hi * 8];
        v16bf bv = load_frag(vp, vp + 16);
        o[di] = wmma_bf16(pf[kc], bv, o[di]);
      }
    }
  }

  // ---- normalize and store attention output as bf16 [B*L, HID] ----
  float rinv[8];
#pragma unroll
  for (int r = 0; r < 8; ++r) rinv[r] = (rl[r] > 0.0f) ? (1.0f / rl[r]) : 0.0f;
#pragma unroll
  for (int di = 0; di < 8; ++di) {
#pragma unroll
    for (int r = 0; r < 8; ++r) {
      const size_t row = (size_t)(b * L_ + qrow0 + hi * 8 + r);
      Ab[row * HID_ + h * D_ + di * 16 + col] = (bf16)(o[di][r] * rinv[r]);
    }
  }
}

// ---------------------------------------------------------------------------
// Host launcher
// ---------------------------------------------------------------------------
extern "C" void kernel_launch(void* const* d_in, const int* in_sizes, int n_in,
                              void* d_out, int out_size, void* d_ws, size_t ws_size,
                              hipStream_t stream) {
  (void)in_sizes; (void)n_in; (void)out_size; (void)ws_size;

  const float* hidden = (const float*)d_in[0];
  const float* Wq     = (const float*)d_in[1];
  const float* Wk     = (const float*)d_in[2];
  const float* Wv     = (const float*)d_in[3];
  const float* Wo     = (const float*)d_in[4];
  const float* qnw    = (const float*)d_in[5];
  const float* knw    = (const float*)d_in[6];
  const float* kbias  = (const float*)d_in[7];
  const float* vbias  = (const float*)d_in[8];
  const int*   pos    = (const int*)d_in[9];
  const unsigned char* amask = (const unsigned char*)d_in[10];
  float* out = (float*)d_out;

  char* ws = (char*)d_ws;
  size_t off = 0;
  auto grab = [&](size_t bytes) -> char* {
    char* p = ws + off;
    off += (bytes + 255) & ~(size_t)255;
    return p;
  };
  const size_t mat  = (size_t)BL_ * HID_ * sizeof(bf16);
  const size_t wmat = (size_t)HID_ * HID_ * sizeof(bf16);

  bf16* Xbf = (bf16*)grab(mat);    // reused as Qh after QKV GEMMs
  bf16* Qb  = (bf16*)grab(mat);    // reused as Ab after post kernel
  bf16* Kb  = (bf16*)grab(mat);
  bf16* Vb  = (bf16*)grab(mat);
  bf16* Kh  = (bf16*)grab(mat);
  bf16* Vh  = (bf16*)grab(mat);
  bf16* Wqb = (bf16*)grab(wmat);
  bf16* Wkb = (bf16*)grab(wmat);
  bf16* Wvb = (bf16*)grab(wmat);
  bf16* Wob = (bf16*)grab(wmat);
  bf16* Qh = Xbf;
  bf16* Ab = Qb;

  cvt_bf16<<<1024, 256, 0, stream>>>(hidden, Xbf, BL_ * HID_);
  cvt_bf16<<<1024, 256, 0, stream>>>(Wq, Wqb, HID_ * HID_);
  cvt_bf16<<<1024, 256, 0, stream>>>(Wk, Wkb, HID_ * HID_);
  cvt_bf16<<<1024, 256, 0, stream>>>(Wv, Wvb, HID_ * HID_);
  cvt_bf16<<<1024, 256, 0, stream>>>(Wo, Wob, HID_ * HID_);

  const dim3 ggrid(HID_ / 256, BL_ / 128);
  gemm_bf16_nt<bf16><<<ggrid, 256, 0, stream>>>(Xbf, Wqb, Qb, FAN_IN);
  gemm_bf16_nt<bf16><<<ggrid, 256, 0, stream>>>(Xbf, Wkb, Kb, FAN_IN);
  gemm_bf16_nt<bf16><<<ggrid, 256, 0, stream>>>(Xbf, Wvb, Vb, FAN_IN);

  post_kernel<<<BL_, 256, 0, stream>>>(Qb, Kb, Vb, qnw, knw, kbias, vbias, pos,
                                       Qh, Kh, Vh);

  attn_kernel<<<dim3(L_ / 128, H_, B_), 256, 0, stream>>>(Qh, Kh, Vh, amask, Ab);

  gemm_bf16_nt<float><<<ggrid, 256, 0, stream>>>(Ab, Wob, out, FAN_IN);
}
